// BlockShufflePermuter_61942018343004
// MI455X (gfx1250) — compile-verified, matrix operand order checked by reading, not run on hardware
//
#include <hip/hip_runtime.h>

#define TEMP_INV 6.6666666666666667f   // 1/0.15
#define CHUNKS 8
#define CSZ 512
#define DIM 4096
#define INTRA_N (CHUNKS*CSZ*CSZ)       // 2097152

typedef _Float16 v16h __attribute__((ext_vector_type(16)));
typedef _Float16 v8h  __attribute__((ext_vector_type(8)));
typedef _Float16 v4h  __attribute__((ext_vector_type(4)));
typedef float    v8f  __attribute__((ext_vector_type(8)));

// ---------------- sinkhorn on 8x8 chunk logits (single tiny block) -------------
__global__ void bsp_chunk_sinkhorn(const float* __restrict__ logits, float* __restrict__ cp) {
    __shared__ float m[64];
    int t = threadIdx.x;            // 0..63
    int i = t >> 3, j = t & 7;
    m[t] = logits[t] * TEMP_INV;
    __syncthreads();
    for (int it = 0; it < 5; ++it) {
        // row logsumexp
        float mx = -1e30f;
        for (int jj = 0; jj < 8; ++jj) mx = fmaxf(mx, m[i*8 + jj]);
        float s = 0.f;
        for (int jj = 0; jj < 8; ++jj) s += __expf(m[i*8 + jj] - mx);
        float lse = mx + __logf(s);
        __syncthreads();
        m[t] -= lse;
        __syncthreads();
        // col logsumexp
        mx = -1e30f;
        for (int ii = 0; ii < 8; ++ii) mx = fmaxf(mx, m[ii*8 + j]);
        s = 0.f;
        for (int ii = 0; ii < 8; ++ii) s += __expf(m[ii*8 + j] - mx);
        lse = mx + __logf(s);
        __syncthreads();
        m[t] -= lse;
        __syncthreads();
    }
    cp[t] = __expf(m[t]);
}

// ---------------- intra sinkhorn: scale, row-norm, col-norm, exp->f16 ----------
__global__ void bsp_scale(const float* __restrict__ src, float* __restrict__ L) {
    int g = blockIdx.x * 256 + threadIdx.x;
    if (g < INTRA_N) L[g] = src[g] * TEMP_INV;
}

__device__ __forceinline__ float bsp_block_reduce_max(float v, float* red) {
    int t = threadIdx.x;
    red[t] = v; __syncthreads();
    for (int s = 128; s > 0; s >>= 1) { if (t < s) red[t] = fmaxf(red[t], red[t+s]); __syncthreads(); }
    float r = red[0]; __syncthreads();
    return r;
}
__device__ __forceinline__ float bsp_block_reduce_sum(float v, float* red) {
    int t = threadIdx.x;
    red[t] = v; __syncthreads();
    for (int s = 128; s > 0; s >>= 1) { if (t < s) red[t] += red[t+s]; __syncthreads(); }
    float r = red[0]; __syncthreads();
    return r;
}

// one block per (chunk,row): 8*512 = 4096 blocks, 256 threads, 2 elems/thread
__global__ void bsp_rownorm(float* __restrict__ L) {
    __shared__ float red[256];
    float* p = L + (size_t)blockIdx.x * CSZ;
    int t = threadIdx.x;
    float v0 = p[t], v1 = p[t + 256];
    float M = bsp_block_reduce_max(fmaxf(v0, v1), red);
    float e = __expf(v0 - M) + __expf(v1 - M);
    float lse = M + __logf(bsp_block_reduce_sum(e, red));
    p[t] = v0 - lse;
    p[t + 256] = v1 - lse;
}

// one block per (chunk,col): stride-512 accesses
__global__ void bsp_colnorm(float* __restrict__ L) {
    __shared__ float red[256];
    int chunk = blockIdx.x >> 9;
    int col   = blockIdx.x & 511;
    float* p = L + (size_t)chunk * CSZ * CSZ + col;
    int t = threadIdx.x;
    float v0 = p[(size_t)t * CSZ], v1 = p[(size_t)(t + 256) * CSZ];
    float M = bsp_block_reduce_max(fmaxf(v0, v1), red);
    float e = __expf(v0 - M) + __expf(v1 - M);
    float lse = M + __logf(bsp_block_reduce_sum(e, red));
    p[(size_t)t * CSZ] = v0 - lse;
    p[(size_t)(t + 256) * CSZ] = v1 - lse;
}

__global__ void bsp_exp_half(const float* __restrict__ L, _Float16* __restrict__ P) {
    int g = blockIdx.x * 256 + threadIdx.x;
    if (g < INTRA_N) P[g] = (_Float16)__expf(L[g]);
}

// ---------------- pass 1: chunk mixing, f32 x -> f16 Y -------------------------
// thread handles 4 consecutive d-positions for all 8 output chunks
__global__ void bsp_mixchunks(const float* __restrict__ x, const float* __restrict__ cp,
                              _Float16* __restrict__ Y, int rows) {
    __shared__ float scp[64];
    if (threadIdx.x < 64) scp[threadIdx.x] = cp[threadIdx.x];
    __syncthreads();
    long g = (long)blockIdx.x * 256 + threadIdx.x;
    if (g >= (long)rows * 128) return;
    long row = g >> 7;
    int d4 = (int)(g & 127) << 2;
    const float* xp = x + row * DIM + d4;
    float xv[8][4];
#pragma unroll
    for (int j = 0; j < 8; ++j) {
        const float4 v = *(const float4*)(xp + j * CSZ);
        xv[j][0] = v.x; xv[j][1] = v.y; xv[j][2] = v.z; xv[j][3] = v.w;
    }
    _Float16* yp = Y + row * DIM + d4;
#pragma unroll
    for (int i = 0; i < 8; ++i) {
        float a0 = 0.f, a1 = 0.f, a2 = 0.f, a3 = 0.f;
#pragma unroll
        for (int j = 0; j < 8; ++j) {
            float w = scp[i*8 + j];
            a0 = fmaf(w, xv[j][0], a0);
            a1 = fmaf(w, xv[j][1], a1);
            a2 = fmaf(w, xv[j][2], a2);
            a3 = fmaf(w, xv[j][3], a3);
        }
        v4h h; h[0] = (_Float16)a0; h[1] = (_Float16)a1; h[2] = (_Float16)a2; h[3] = (_Float16)a3;
        *(v4h*)(yp + i * CSZ) = h;
    }
}

// ---------------- pass 2: WMMA GEMM per chunk: Out_c = Y_c * P_c^T -------------
// A fragment (16x32 f16): lane m holds row m; K 0..7 & 16..23 (lanes 0-15),
// K 8..15 & 24..31 (lanes 16-31) -> two B128 loads at +0 and +16 halves.
// B fragment (32x16 f16): lane n holds column n of B == row n of P (K-contiguous),
// same two-B128 pattern. P row-major f16 gives this directly.
__device__ __forceinline__ v16h bsp_frag(const _Float16* p) {
    v8h lo = *(const v8h*)p;
    v8h hi = *(const v8h*)(p + 16);
    return __builtin_shufflevector(lo, hi, 0,1,2,3,4,5,6,7,8,9,10,11,12,13,14,15);
}

__global__ void __launch_bounds__(256)
bsp_gemm_intra(const _Float16* __restrict__ Y, const _Float16* __restrict__ P,
               float* __restrict__ out, int rows) {
    const int lane = threadIdx.x & 31;
    const int wid  = threadIdx.x >> 5;
    const int wm   = wid & 3;           // 4 waves along M (32 rows each)
    const int wn   = wid >> 2;          // 2 waves along N (64 cols each)
    const int c    = blockIdx.z;
    const long mBase = (long)blockIdx.x * 128 + wm * 32;
    const int  nBase = blockIdx.y * 128 + wn * 64;
    const int  lr = lane & 15;
    const int  lh = lane >> 4;          // which K-half this lane loads

    const _Float16* Pc = P + (size_t)c * CSZ * CSZ;
    const _Float16* aRow0 = Y + (mBase + lr) * (long)DIM + c * CSZ + lh * 8;
    const _Float16* aRow1 = aRow0 + 16L * DIM;
    const _Float16* bRow[4];
#pragma unroll
    for (int t = 0; t < 4; ++t)
        bRow[t] = Pc + (size_t)(nBase + 16*t + lr) * CSZ + lh * 8;

    v8f acc[2][4];
#pragma unroll
    for (int i = 0; i < 2; ++i)
#pragma unroll
        for (int t = 0; t < 4; ++t) acc[i][t] = (v8f)(0.f);

    for (int k = 0; k < CSZ; k += 32) {
        v16h a0 = bsp_frag(aRow0 + k);
        v16h a1 = bsp_frag(aRow1 + k);
        v16h b[4];
#pragma unroll
        for (int t = 0; t < 4; ++t) b[t] = bsp_frag(bRow[t] + k);
#pragma unroll
        for (int t = 0; t < 4; ++t) {
            acc[0][t] = __builtin_amdgcn_wmma_f32_16x16x32_f16(
                false, a0, false, b[t], (short)0, acc[0][t], false, false);
            acc[1][t] = __builtin_amdgcn_wmma_f32_16x16x32_f16(
                false, a1, false, b[t], (short)0, acc[1][t], false, false);
        }
    }

    // C/D layout: VGPR r, lanes 0-15 -> (M=r, N=lane); lanes 16-31 -> (M=r+8, N=lane-16)
    const int ocol = c * CSZ + nBase;
#pragma unroll
    for (int i = 0; i < 2; ++i) {
#pragma unroll
        for (int t = 0; t < 4; ++t) {
            float* op = out + (mBase + 16*i + lh*8) * (long)DIM + ocol + 16*t + lr;
#pragma unroll
            for (int r = 0; r < 8; ++r)
                op[(long)r * DIM] = acc[i][t][r];
        }
    }
}

extern "C" void kernel_launch(void* const* d_in, const int* in_sizes, int n_in,
                              void* d_out, int out_size, void* d_ws, size_t ws_size,
                              hipStream_t stream) {
    const float* x            = (const float*)d_in[0];
    const float* chunk_logits = (const float*)d_in[1];
    const float* intra_logits = (const float*)d_in[2];
    float* out = (float*)d_out;

    char* ws = (char*)d_ws;
    float*    cp = (float*)ws;                                   // 256 B
    float*    L  = (float*)(ws + 4096);                          // 8 MB f32 log_p
    _Float16* Pf = (_Float16*)(ws + 4096 + (8u << 20));          // 4 MB f16 intra_perm
    _Float16* Yf = (_Float16*)(ws + 4096 + (12u << 20));         // 128 MB f16 mixed x

    const int rows = in_sizes[0] / DIM;                          // 16384

    bsp_chunk_sinkhorn<<<1, 64, 0, stream>>>(chunk_logits, cp);
    bsp_scale<<<INTRA_N / 256, 256, 0, stream>>>(intra_logits, L);
    for (int it = 0; it < 5; ++it) {
        bsp_rownorm<<<CHUNKS * CSZ, 256, 0, stream>>>(L);
        bsp_colnorm<<<CHUNKS * CSZ, 256, 0, stream>>>(L);
    }
    bsp_exp_half<<<INTRA_N / 256, 256, 0, stream>>>(L, Pf);

    bsp_mixchunks<<<(rows * 128) / 256, 256, 0, stream>>>(x, cp, Yf, rows);

    dim3 grid(rows / 128, CSZ / 128, CHUNKS);
    bsp_gemm_intra<<<grid, 256, 0, stream>>>(Yf, Pf, out, rows);
}
